// HSAAttention_40089224741578
// MI455X (gfx1250) — compile-verified, matrix operand order checked by reading, not run on hardware
//
#include <hip/hip_runtime.h>

typedef __bf16 bf16;
typedef __attribute__((ext_vector_type(16))) __bf16 v16bf;
typedef __attribute__((ext_vector_type(8)))  __bf16 v8bf;
typedef __attribute__((ext_vector_type(8)))  unsigned short v8us;
typedef __attribute__((ext_vector_type(4)))  unsigned short v4us;
typedef __attribute__((ext_vector_type(8)))  float  v8f;

#define WMMA_BF16(a, b, c) \
  __builtin_amdgcn_wmma_f32_16x16x32_bf16(false, (a), false, (b), (short)0, (c), false, false)

// Lane fragment load from an LDS row (row-major, stride in elements).
// CDNA5 K-map: VGPR0..3 hold K = hi8+0..7, VGPR4..7 hold K = 16+hi8+0..7,
// hi8 = (lane>=16)*8. Two ds_load_b128.
__device__ __forceinline__ v16bf frag_load(const bf16* rowp, int lh8) {
  v8bf lo = *(const v8bf*)(rowp + lh8);
  v8bf hi = *(const v8bf*)(rowp + 16 + lh8);
  v16bf r;
#pragma unroll
  for (int i = 0; i < 8; ++i) { r[i] = lo[i]; r[i + 8] = hi[i]; }
  return r;
}

__device__ __forceinline__ v8bf pack8(const float4& f0, const float4& f1) {
  v8bf v;
  v[0] = (bf16)f0.x; v[1] = (bf16)f0.y; v[2] = (bf16)f0.z; v[3] = (bf16)f0.w;
  v[4] = (bf16)f1.x; v[5] = (bf16)f1.y; v[6] = (bf16)f1.z; v[7] = (bf16)f1.w;
  return v;
}

// ---------------------------------------------------------------------------
// NT GEMM: C[M,N] = X[M,K] @ W[N,K]^T (+bias) (*row_scale), batched.
// X f32 or bf16 (bf16-converted in LDS), W f32. 256x128x32 block, 8 wave32s,
// 64x64 wave tile => 16 v_wmma per K-step. Register-staged, double-buffered
// LDS pipeline: global prefetch of tile k+1 overlaps the WMMAs of tile k,
// one barrier per K-step.
// ---------------------------------------------------------------------------
template <typename XT>
__global__ __launch_bounds__(256) void gemm_nt(
    const XT* __restrict__ X, long xstride,
    const float* __restrict__ W, long wstride,
    const float* __restrict__ bias,
    const float* __restrict__ rscale, long rstride,
    float* __restrict__ Cf, bf16* __restrict__ Cb, long cstride,
    int M, int N, int K) {
  constexpr int BM = 256, BN = 128, BK = 32, LDT = BK + 8;
  __shared__ bf16 As[2][BM * LDT];
  __shared__ bf16 Bs[2][BN * LDT];

  const int bz = blockIdx.z;
  X += (long)bz * xstride;
  W += (long)bz * wstride;
  if (Cf) Cf += (long)bz * cstride;
  if (Cb) Cb += (long)bz * cstride;
  if (rscale) rscale += (long)bz * rstride;

  const int bm = blockIdx.y * BM, bn = blockIdx.x * BN;
  const int tid = threadIdx.x, lane = tid & 31, wv = tid >> 5;
  const int wm = (wv & 3) * 64, wn = (wv >> 2) * 64;
  const int lc = lane & 15, lh8 = (lane >> 4) * 8;

  // staging pointers (per-thread fixed rows)
  const XT* aptr = X + (long)(bm + tid) * K;             // one 32-wide A row
  const int br = tid >> 1, bkh = (tid & 1) * 16;
  const bool bvalid = (bn + br) < N;
  const float* bptr = W + (long)(bn + br) * K + bkh;     // half a W row

  // prefetch registers
  float4 aregf[8];  // XT == float
  uint4  aregb[4];  // XT == bf16
  float4 breg[4];

  auto load_tile = [&](int k0) {
    if constexpr (sizeof(XT) == 4) {
      const float4* s = reinterpret_cast<const float4*>(aptr + k0);
#pragma unroll
      for (int c = 0; c < 8; ++c) aregf[c] = s[c];
    } else {
      const uint4* s = reinterpret_cast<const uint4*>(aptr + k0);
#pragma unroll
      for (int c = 0; c < 4; ++c) aregb[c] = s[c];
    }
    if (bvalid) {
      const float4* s = reinterpret_cast<const float4*>(bptr + k0);
#pragma unroll
      for (int c = 0; c < 4; ++c) breg[c] = s[c];
    } else {
      float4 z = {0.f, 0.f, 0.f, 0.f};
#pragma unroll
      for (int c = 0; c < 4; ++c) breg[c] = z;
    }
  };
  auto store_tile = [&](int p) {
    bf16* da = As[p] + tid * LDT;
    if constexpr (sizeof(XT) == 4) {
#pragma unroll
      for (int c = 0; c < 4; ++c)
        *reinterpret_cast<v8bf*>(da + c * 8) = pack8(aregf[2 * c], aregf[2 * c + 1]);
    } else {
#pragma unroll
      for (int c = 0; c < 4; ++c)
        *reinterpret_cast<uint4*>(da + c * 8) = aregb[c];
    }
    bf16* db = Bs[p] + br * LDT + bkh;
#pragma unroll
    for (int c = 0; c < 2; ++c)
      *reinterpret_cast<v8bf*>(db + c * 8) = pack8(breg[2 * c], breg[2 * c + 1]);
  };

  v8f acc[4][4] = {};

  const int nk = K / BK;
  load_tile(0);
  store_tile(0);
  __syncthreads();

  for (int ki = 0; ki < nk; ++ki) {
    const int p = ki & 1;
    if (ki + 1 < nk) load_tile((ki + 1) * BK);  // prefetch next tile

    v16bf af[4], bfv[4];
#pragma unroll
    for (int fm = 0; fm < 4; ++fm)
      af[fm] = frag_load(As[p] + (wm + fm * 16 + lc) * LDT, lh8);
#pragma unroll
    for (int fn = 0; fn < 4; ++fn)
      bfv[fn] = frag_load(Bs[p] + (wn + fn * 16 + lc) * LDT, lh8);
#pragma unroll
    for (int fm = 0; fm < 4; ++fm)
#pragma unroll
      for (int fn = 0; fn < 4; ++fn)
        acc[fm][fn] = WMMA_BF16(af[fm], bfv[fn], acc[fm][fn]);

    if (ki + 1 < nk) store_tile(1 - p);  // fill other buffer after compute
    __syncthreads();
  }

  // epilogue: D layout -> row = j + 8*(lane>=16), col = lane&15
  const int col0 = bn + wn + lc;
  float bias_v[4];
#pragma unroll
  for (int fn = 0; fn < 4; ++fn)
    bias_v[fn] = (bias && col0 + fn * 16 < N) ? bias[col0 + fn * 16] : 0.0f;
#pragma unroll
  for (int fm = 0; fm < 4; ++fm) {
#pragma unroll
    for (int j = 0; j < 8; ++j) {
      int gm = bm + wm + fm * 16 + (lane >> 4) * 8 + j;
      float rs = rscale ? rscale[gm] : 1.0f;
      long rowoff = (long)gm * N + col0;
#pragma unroll
      for (int fn = 0; fn < 4; ++fn) {
        if (col0 + fn * 16 >= N) continue;
        float v = (acc[fm][fn][j] + bias_v[fn]) * rs;
        if (Cf) Cf[rowoff + fn * 16] = v;
        if (Cb) Cb[rowoff + fn * 16] = (bf16)v;
      }
    }
  }
}

// ---------------------------------------------------------------------------
// TN GEMM with split-K atomics: Mt[b,d,n] += sum_s V[b,s,d] * G[b,s,n]
// (Mt = V^T G per batch, NS=64 compile-time). LDS transpose staged by packing
// two s-rows per u32 (ds_store_b32). Block 128(d) x 64(n).
// ---------------------------------------------------------------------------
__global__ __launch_bounds__(256) void gemm_tn_atomic(
    const bf16* __restrict__ V, const bf16* __restrict__ G,
    float* __restrict__ Mt, int S, int D, int kchunk) {
  constexpr int NS = 64;
  constexpr int BD = 128, BK = 32, LDT = BK + 8;
  __shared__ bf16 Xs[BD * LDT];  // [d][s]
  __shared__ bf16 Ys[NS * LDT];  // [n][s]

  const int bb = blockIdx.z;
  const int d0 = blockIdx.x * BD;
  const int s_begin = blockIdx.y * kchunk;
  const bf16* Vb = V + (long)bb * S * D;
  const bf16* Gb = G + (long)bb * S * NS;
  float* Mb = Mt + (long)bb * D * NS;

  const int tid = threadIdx.x, lane = tid & 31, wv = tid >> 5;
  const int wd = (wv & 3) * 32, wn = (wv >> 2) * 32;
  const int lc = lane & 15, lh8 = (lane >> 4) * 8;
  const int sp = (tid & 15) * 2;  // s-pair base for staging

  v8f acc[2][2] = {};

  for (int s0 = s_begin; s0 < s_begin + kchunk; s0 += BK) {
    {  // V tile [32 s][128 d] -> Xs[d][s], 2 s-rows packed per u32 store
      int dj = (tid >> 4) * 8;
      const bf16* r0 = Vb + (long)(s0 + sp) * D + d0 + dj;
      v8us a = *(const v8us*)r0;
      v8us b = *(const v8us*)(r0 + D);
#pragma unroll
      for (int i = 0; i < 8; ++i) {
        unsigned int w = (unsigned int)a[i] | ((unsigned int)b[i] << 16);
        *reinterpret_cast<unsigned int*>(Xs + (dj + i) * LDT + sp) = w;
      }
    }
    {  // G tile [32 s][64 n] -> Ys[n][s]
      int nj = (tid >> 4) * 4;
      const bf16* r0 = Gb + (long)(s0 + sp) * NS + nj;
      v4us a = *(const v4us*)r0;
      v4us b = *(const v4us*)(r0 + NS);
#pragma unroll
      for (int i = 0; i < 4; ++i) {
        unsigned int w = (unsigned int)a[i] | ((unsigned int)b[i] << 16);
        *reinterpret_cast<unsigned int*>(Ys + (nj + i) * LDT + sp) = w;
      }
    }
    __syncthreads();

    v16bf af[2], bfv[2];
#pragma unroll
    for (int fm = 0; fm < 2; ++fm)
      af[fm] = frag_load(Xs + (wd + fm * 16 + lc) * LDT, lh8);
#pragma unroll
    for (int fn = 0; fn < 2; ++fn)
      bfv[fn] = frag_load(Ys + (wn + fn * 16 + lc) * LDT, lh8);
#pragma unroll
    for (int fm = 0; fm < 2; ++fm)
#pragma unroll
      for (int fn = 0; fn < 2; ++fn)
        acc[fm][fn] = WMMA_BF16(af[fm], bfv[fn], acc[fm][fn]);
    __syncthreads();
  }

  // single base pointer; all 32 atomics get compile-time element offsets
  float* base = Mb + (long)(d0 + wd + (lane >> 4) * 8) * NS + wn + lc;
#pragma unroll
  for (int fm = 0; fm < 2; ++fm)
#pragma unroll
    for (int fn = 0; fn < 2; ++fn)
#pragma unroll
      for (int j = 0; j < 8; ++j)
        atomicAdd(base + ((fm * 16 + j) * NS + fn * 16), acc[fm][fn][j]);
}

// ---------------------------------------------------------------------------
// Small helper kernels
// ---------------------------------------------------------------------------
__global__ void zero_f32(float* p, int n) {
  int i = blockIdx.x * 256 + threadIdx.x;
  if (i < n) p[i] = 0.0f;
}

__global__ __launch_bounds__(256) void row_sumsq(const float* __restrict__ X,
                                                 float* __restrict__ out, int C) {
  const float* p = X + (long)blockIdx.x * C;
  float s = 0.0f;
  for (int i = threadIdx.x; i < C; i += 256) { float v = p[i]; s += v * v; }
  __shared__ float red[256];
  red[threadIdx.x] = s;
  __syncthreads();
  for (int off = 128; off; off >>= 1) {
    if (threadIdx.x < off) red[threadIdx.x] += red[threadIdx.x + off];
    __syncthreads();
  }
  if (threadIdx.x == 0) out[blockIdx.x] = red[0];
}

// G = exp(-(t2 - 2*cross + c2) * 0.5*exp(-2*log_scale)); emit G(bf16),
// G*amp(bf16), accumulate colsum[b,n] = sum_s G. 256 thr = 4 s x 64 n.
__global__ __launch_bounds__(256) void splat_g(
    const float* __restrict__ cross, const float* __restrict__ t2,
    const float* __restrict__ c2, const float* __restrict__ lsc,
    const float* __restrict__ amp, bf16* __restrict__ Gb,
    bf16* __restrict__ Gwb, float* __restrict__ colsum, int S) {
  constexpr int NS = 64;
  __shared__ float part[256];
  int t = threadIdx.x, n = t & 63, sl = t >> 6;
  int spb = S >> 2;
  int b = blockIdx.x / spb;
  int s = (blockIdx.x % spb) * 4 + sl;
  long idx = ((long)b * S + s) * NS + n;
  float itv = 0.5f * __expf(-2.0f * lsc[n]);
  float d2 = t2[b * S + s] - 2.0f * cross[idx] + c2[n];
  float g = __expf(-d2 * itv);
  Gb[idx] = (bf16)g;
  Gwb[idx] = (bf16)(g * amp[n]);
  part[t] = g;
  __syncthreads();
  if (sl == 0)
    atomicAdd(&colsum[b * NS + n],
              part[n] + part[64 + n] + part[128 + n] + part[192 + n]);
}

// inv_row[b,s] = 1 / (sum_n Gw[b,s,n]*colsum[b,n] + eps)
__global__ void row_norm(const bf16* __restrict__ Gw,
                         const float* __restrict__ colsum,
                         float* __restrict__ invrow, int S) {
  constexpr int NS = 64;
  int i = blockIdx.x * 256 + threadIdx.x;  // over B*S
  int b = i / S;
  const bf16* g = Gw + (long)i * NS;
  const float* cs = colsum + b * NS;
  float acc = 0.0f;
#pragma unroll 8
  for (int n = 0; n < NS; ++n) acc += (float)g[n] * cs[n];
  invrow[i] = 1.0f / (acc + 1e-8f);
}

// ---------------------------------------------------------------------------
extern "C" void kernel_launch(void* const* d_in, const int* in_sizes, int n_in,
                              void* d_out, int out_size, void* d_ws,
                              size_t ws_size, hipStream_t stream) {
  constexpr int B = 4, S = 2048, D = 1024, NS = 64, BS = B * S;

  const float* query   = (const float*)d_in[0];
  const float* value   = (const float*)d_in[2];   // d_in[1] = key (k_proj discarded)
  const float* Wq      = (const float*)d_in[3];
  const float* bq      = (const float*)d_in[4];
  const float* Wv      = (const float*)d_in[7];   // d_in[5,6] = Wk,bk (unused)
  const float* bv      = (const float*)d_in[8];
  const float* Wo      = (const float*)d_in[9];
  const float* bo      = (const float*)d_in[10];
  const float* centers = (const float*)d_in[11];
  const float* lsc     = (const float*)d_in[12];
  const float* amp     = (const float*)d_in[13];
  float* out = (float*)d_out;

  char* p = (char*)d_ws;
  float* qf32   = (float*)p;  p += (size_t)BS * D * 4;
  bf16*  v_bf   = (bf16*)p;   p += (size_t)BS * D * 2;
  float* cross  = (float*)p;  p += (size_t)BS * NS * 4;
  float* t2     = (float*)p;  p += (size_t)BS * 4;
  float* c2     = (float*)p;  p += (size_t)NS * 4;
  bf16*  G_bf   = (bf16*)p;   p += (size_t)BS * NS * 2;
  bf16*  Gw_bf  = (bf16*)p;   p += (size_t)BS * NS * 2;
  float* colsum = (float*)p;  p += (size_t)B * NS * 4;      // zeroed
  float* Mt     = (float*)p;  p += (size_t)B * D * NS * 4;  // zeroed (adjacent)
  float* invrow = (float*)p;  p += (size_t)BS * 4;
  bf16*  outpre = (bf16*)p;   p += (size_t)BS * D * 2;

  dim3 blk(256);
  const int nZero = B * NS + B * D * NS;
  zero_f32<<<(nZero + 255) / 256, blk, 0, stream>>>(colsum, nZero);

  // q = query @ Wq^T + bq -> f32
  gemm_nt<float><<<dim3(D / 128, BS / 256, 1), blk, 0, stream>>>(
      query, 0, Wq, 0, bq, nullptr, 0, qf32, nullptr, 0, BS, D, D);
  // v = value @ Wv^T + bv -> bf16
  gemm_nt<float><<<dim3(D / 128, BS / 256, 1), blk, 0, stream>>>(
      value, 0, Wv, 0, bv, nullptr, 0, nullptr, v_bf, 0, BS, D, D);
  // cross = q @ centers^T -> f32 (N=64)
  gemm_nt<float><<<dim3(1, BS / 256, 1), blk, 0, stream>>>(
      qf32, 0, centers, 0, nullptr, nullptr, 0, cross, nullptr, 0, BS, NS, D);

  row_sumsq<<<BS, blk, 0, stream>>>(qf32, t2, D);
  row_sumsq<<<NS, blk, 0, stream>>>(centers, c2, D);

  splat_g<<<B * (S / 4), blk, 0, stream>>>(cross, t2, c2, lsc, amp, G_bf,
                                           Gw_bf, colsum, S);

  // Mt[b] = V[b]^T @ G[b] (split-K over s: 8 chunks of 256)
  gemm_tn_atomic<<<dim3(D / 128, 8, B), blk, 0, stream>>>(v_bf, G_bf, Mt, S, D,
                                                          S / 8);

  row_norm<<<BS / 256, blk, 0, stream>>>(Gw_bf, colsum, invrow, S);

  // out_pre[b] = (Gw[b] @ Mt[b]^T) * invrow -> bf16 (batched, K=64)
  gemm_nt<bf16><<<dim3(D / 128, S / 256, B), blk, 0, stream>>>(
      Gw_bf, (long)S * NS, Mt, (long)D * NS, nullptr, invrow, S, nullptr,
      outpre, (long)S * D, S, D, NS);

  // out = out_pre @ Wo^T + bo -> f32 d_out
  gemm_nt<bf16><<<dim3(D / 128, BS / 256, 1), blk, 0, stream>>>(
      outpre, 0, Wo, 0, bo, nullptr, 0, out, nullptr, 0, BS, D, D);

  (void)in_sizes; (void)n_in; (void)out_size; (void)ws_size;
}
